// ClassicalSelfAttention_65481071400334
// MI455X (gfx1250) — matrix operand order repaired and checked by reading, not compiled, not run on hardware
//
#include <hip/hip_runtime.h>
#include <hip/hip_bf16.h>

#define DEV __device__ __forceinline__

typedef _Float16 half8  __attribute__((ext_vector_type(8)));
typedef _Float16 half16 __attribute__((ext_vector_type(16)));
typedef float    f32x8  __attribute__((ext_vector_type(8)));
typedef unsigned int uint32x4 __attribute__((ext_vector_type(4)));
typedef int      int32x4 __attribute__((ext_vector_type(4)));
typedef int      int32x8 __attribute__((ext_vector_type(8)));

static constexpr int Bsz = 4;
static constexpr int Lsz = 2048;
static constexpr int Dsz = 1024;
static constexpr int Hn  = 16;
static constexpr int Hd  = 64;
static constexpr int Msz = Bsz * Lsz;   // 8192 rows

#if defined(__has_builtin)
#if __has_builtin(__builtin_amdgcn_tensor_load_to_lds) && \
    __has_builtin(__builtin_amdgcn_s_wait_tensorcnt)
#define USE_TDM 1
#endif
#endif
#ifndef USE_TDM
#define USE_TDM 0
#endif

DEV half16 frag_cat(half8 c0, half8 c1) {
  return __builtin_shufflevector(c0, c1, 0,1,2,3,4,5,6,7,8,9,10,11,12,13,14,15);
}

// A operand (16x32 f16). lane m = lane&15; lanes<16 hold K 0..15,
// lanes>=16 hold K 16..31  -> two 16B chunks at +hi*16 and +hi*16+8.
DEV half16 load_frag_a(const _Float16* base, int ld) {
  const int lane = threadIdx.x & 31;
  const int m  = lane & 15;
  const int hi = lane >> 4;
  const _Float16* p = base + (size_t)m * ld + hi * 16;
  half8 c0 = *(const half8*)(p);
  half8 c1 = *(const half8*)(p + 8);
  return frag_cat(c0, c1);
}

// B operand (32x16 f16). lane n = lane&15 holds column n; lanes<16 hold
// K 0..15, lanes>=16 hold K 16..31 (32 contiguous bytes per lane).
DEV half16 load_frag_b(const _Float16* base, int ld) {
  const int lane = threadIdx.x & 31;
  const int n  = lane & 15;
  const int hi = lane >> 4;
  const _Float16* p = base + (size_t)n * ld + hi * 16;
  half8 c0 = *(const half8*)(p);
  half8 c1 = *(const half8*)(p + 8);
  return frag_cat(c0, c1);
}

DEV f32x8 wmma16(half16 a, half16 b, f32x8 c) {
  // (neg_a, A, neg_b, B, c_mod, C, reuse_a, reuse_b)
  return __builtin_amdgcn_wmma_f32_16x16x32_f16(false, a, false, b, (short)0, c,
                                                false, false);
}

#if USE_TDM
// Issue one TDM 2-D tile load (tile origin == tensor origin, so no OOB).
// D# layout per CDNA5 ISA 8.3/8.4. data_size = 2 bytes (f16).
DEV void tdm_load_2d_f16(unsigned lds_addr, const void* gsrc,
                         unsigned tensor_d0, unsigned tensor_d1,
                         unsigned tile_d0, unsigned tile_d1,
                         unsigned stride0_elems) {
  const unsigned long long ga = (unsigned long long)(size_t)gsrc;
  uint32x4 g0;
  g0.x = 1u;                                        // count=1, no gather
  g0.y = lds_addr;                                  // LDS byte address
  g0.z = (unsigned)(ga & 0xffffffffull);            // global_addr[31:0]
  g0.w = (unsigned)((ga >> 32) & 0x01ffffffull)     // global_addr[56:32]
         | (2u << 30);                              // type=2 ("image")
  int32x8 g1;
  g1[0] = (int)(1u << 16);                          // data_size=1 -> 2 bytes
  g1[1] = (int)((tensor_d0 & 0xffffu) << 16);       // tensor_dim0[15:0]
  g1[2] = (int)((tensor_d0 >> 16) |                 // tensor_dim0[31:16]
                ((tensor_d1 & 0xffffu) << 16));     // tensor_dim1[15:0]
  g1[3] = (int)((tensor_d1 >> 16) |                 // tensor_dim1[31:16]
                (tile_d0 << 16));                   // tile_dim0
  g1[4] = (int)tile_d1;                             // tile_dim1 (tile_dim2=0)
  g1[5] = (int)stride0_elems;                       // tensor_dim0_stride[31:0]
  g1[6] = 0;                                        // stride0 hi / stride1 lo
  g1[7] = 0;                                        // stride1 hi
  const int32x4 z4 = {0, 0, 0, 0};
#if defined(__clang_major__) && __clang_major__ >= 23
  const int32x8 z8 = {0, 0, 0, 0, 0, 0, 0, 0};
  __builtin_amdgcn_tensor_load_to_lds(g0, g1, z4, z4, z8, 0);
#else
  __builtin_amdgcn_tensor_load_to_lds(g0, g1, z4, z4, 0);
#endif
}
#endif

__global__ void f32_to_f16_kernel(const float* __restrict__ src,
                                  _Float16* __restrict__ dst, int nElem) {
  int i = blockIdx.x * blockDim.x + threadIdx.x;
  const int stride = gridDim.x * blockDim.x;
  for (; i < nElem; i += stride) dst[i] = (_Float16)src[i];
}

// ---------------------------------------------------------------------------
// Fused QKV projection: C = X * W^T + b.  blockIdx.z selects Q / K / V.
// Block = 8 waves (2x4), each wave a 64x64 tile (4x4 WMMA accumulators).
// Q,K stored f16 as [b,h,l,hd]; V stored f16 transposed [b,h,hd,l].
// ---------------------------------------------------------------------------
__global__ __launch_bounds__(256)
void qkv_gemm_kernel(const _Float16* __restrict__ Xh,
                     const _Float16* __restrict__ Wq,
                     const _Float16* __restrict__ Wk,
                     const _Float16* __restrict__ Wv,
                     const float* __restrict__ bq,
                     const float* __restrict__ bk,
                     const float* __restrict__ bv,
                     _Float16* __restrict__ Qh,
                     _Float16* __restrict__ Kh,
                     _Float16* __restrict__ Vt) {
  const int wave = threadIdx.x >> 5;
  const int lane = threadIdx.x & 31;
  const int wm = wave >> 2;                 // 0..1
  const int wn = wave & 3;                  // 0..3
  const int row0 = blockIdx.y * 128 + wm * 64;
  const int col0 = blockIdx.x * 256 + wn * 64;
  const int z = blockIdx.z;
  const _Float16* W = (z == 0) ? Wq : (z == 1) ? Wk : Wv;
  const float* bias = (z == 0) ? bq : (z == 1) ? bk : bv;

  const f32x8 zero = {0.f,0.f,0.f,0.f,0.f,0.f,0.f,0.f};
  f32x8 acc[4][4];
  #pragma unroll
  for (int i = 0; i < 4; ++i)
    #pragma unroll
    for (int j = 0; j < 4; ++j) acc[i][j] = zero;

  for (int k0 = 0; k0 < Dsz; k0 += 32) {
    half16 a[4], bfr[4];
    #pragma unroll
    for (int i = 0; i < 4; ++i)
      a[i] = load_frag_a(Xh + (size_t)(row0 + i * 16) * Dsz + k0, Dsz);
    #pragma unroll
    for (int j = 0; j < 4; ++j)
      bfr[j] = load_frag_b(W + (size_t)(col0 + j * 16) * Dsz + k0, Dsz);
    #pragma unroll
    for (int i = 0; i < 4; ++i)
      #pragma unroll
      for (int j = 0; j < 4; ++j)
        acc[i][j] = wmma16(a[i], bfr[j], acc[i][j]);
  }

  const int n  = lane & 15;
  const int hi = lane >> 4;
  _Float16* QKdst = (z == 0) ? Qh : Kh;
  #pragma unroll
  for (int j = 0; j < 4; ++j) {
    const int col = col0 + j * 16 + n;
    const float bvv = bias[col];
    const int h  = col >> 6;
    const int hd = col & 63;
    #pragma unroll
    for (int i = 0; i < 4; ++i) {
      #pragma unroll
      for (int r = 0; r < 8; ++r) {
        const int row = row0 + i * 16 + r + 8 * hi;
        const float v = acc[i][j][r] + bvv;
        const int bb = row >> 11;          // / L
        const int ll = row & (Lsz - 1);
        if (z == 2)
          Vt[(((size_t)bb * Hn + h) * Hd + hd) * Lsz + ll] = (_Float16)v;
        else
          QKdst[(((size_t)bb * Hn + h) * Lsz + ll) * Hd + hd] = (_Float16)v;
      }
    }
  }
}

// ---------------------------------------------------------------------------
// Flash attention: block = 8 waves, each wave owns 16 query rows (block: 128).
// K/V 64-key tiles are staged into LDS by the Tensor Data Mover (wave 0,
// double-buffered behind TENSORcnt) so the 8 waves share one fetch.
// Per tile: S = Q*K^T (WMMA), online softmax (shfl_xor row reduce),
// P spilled f16 via per-wave LDS (C-layout -> A-layout), O += P*V (WMMA).
// ---------------------------------------------------------------------------
__global__ __launch_bounds__(256)
void flash_attn_kernel(const _Float16* __restrict__ Qh,
                       const _Float16* __restrict__ Kh,
                       const _Float16* __restrict__ Vt,
                       _Float16* __restrict__ Oh) {
  __shared__ alignas(16) _Float16 shK[2][64 * 64];   // [key][hd]
  __shared__ alignas(16) _Float16 shV[2][64 * 64];   // [hd][key]
  __shared__ alignas(16) _Float16 shP[8][16 * 64];   // per-wave P spill
  const int wave = threadIdx.x >> 5;
  const int lane = threadIdx.x & 31;
  const int n  = lane & 15;
  const int hi = lane >> 4;
  const int b = blockIdx.z, h = blockIdx.y;
  const _Float16* Qb = Qh + ((size_t)(b * Hn + h) * Lsz) * Hd;
  const _Float16* Kb = Kh + ((size_t)(b * Hn + h) * Lsz) * Hd;
  const _Float16* Vb = Vt + ((size_t)(b * Hn + h) * Hd) * Lsz;
  _Float16* pl = shP[wave];
  const int q0 = blockIdx.x * 128 + wave * 16;

  const half16 aQ0 = load_frag_a(Qb + (size_t)q0 * Hd + 0,  Hd);
  const half16 aQ1 = load_frag_a(Qb + (size_t)q0 * Hd + 32, Hd);

  const f32x8 zero = {0.f,0.f,0.f,0.f,0.f,0.f,0.f,0.f};
  f32x8 o[4] = {zero, zero, zero, zero};
  float mrow[8], lrow[8];
  #pragma unroll
  for (int r = 0; r < 8; ++r) { mrow[r] = -1e30f; lrow[r] = 0.0f; }

  const float scale = 0.125f;               // 1/sqrt(Hd)

#if USE_TDM
  const unsigned aK[2] = {(unsigned)(size_t)&shK[0][0], (unsigned)(size_t)&shK[1][0]};
  const unsigned aV[2] = {(unsigned)(size_t)&shV[0][0], (unsigned)(size_t)&shV[1][0]};
  if (wave == 0) {   // prologue: tile 0 -> buffer 0
    tdm_load_2d_f16(aK[0], Kb, Hd, 64, Hd, 64, Hd);
    tdm_load_2d_f16(aV[0], Vb, 64, Hd, 64, Hd, Lsz);
  }
#endif

  for (int key0 = 0; key0 < Lsz; key0 += 64) {
#if USE_TDM
    const int buf = (key0 >> 6) & 1;
    if (wave == 0) {
      if (key0 + 64 < Lsz) {   // kick off next tile into the other buffer
        const int nb = buf ^ 1;
        tdm_load_2d_f16(aK[nb], Kb + (size_t)(key0 + 64) * Hd, Hd, 64, Hd, 64, Hd);
        tdm_load_2d_f16(aV[nb], Vb + (key0 + 64), 64, Hd, 64, Hd, Lsz);
        __builtin_amdgcn_s_wait_tensorcnt(2);   // drain current tile only
      } else {
        __builtin_amdgcn_s_wait_tensorcnt(0);
      }
    }
    __syncthreads();                            // tile `buf` visible to all
#else
    const int buf = 0;
    __syncthreads();                            // previous tile fully consumed
    {
      const half8* srcK = (const half8*)(Kb + (size_t)key0 * Hd);
      half8* dstK = (half8*)shK[0];
      half8* dstV = (half8*)shV[0];
      for (int i = threadIdx.x; i < 512; i += 256) {
        dstK[i] = srcK[i];
        const int r = i >> 3, c = i & 7;
        dstV[i] = *(const half8*)(Vb + (size_t)r * Lsz + key0 + c * 8);
      }
    }
    __syncthreads();
#endif
    const _Float16* Kt = shK[buf];
    const _Float16* Vl = shV[buf];

    f32x8 s[4];
    #pragma unroll
    for (int t = 0; t < 4; ++t) {
      const half16 bK0 = load_frag_b(Kt + (t * 16) * Hd + 0,  Hd);
      const half16 bK1 = load_frag_b(Kt + (t * 16) * Hd + 32, Hd);
      f32x8 a = zero;
      a = wmma16(aQ0, bK0, a);
      a = wmma16(aQ1, bK1, a);
      #pragma unroll
      for (int r = 0; r < 8; ++r) s[t][r] = a[r] * scale;
    }
    float alpha[8];
    #pragma unroll
    for (int r = 0; r < 8; ++r) {
      float v = fmaxf(fmaxf(s[0][r], s[1][r]), fmaxf(s[2][r], s[3][r]));
      v = fmaxf(v, __shfl_xor(v, 1, 32));
      v = fmaxf(v, __shfl_xor(v, 2, 32));
      v = fmaxf(v, __shfl_xor(v, 4, 32));
      v = fmaxf(v, __shfl_xor(v, 8, 32));
      const float nm = fmaxf(mrow[r], v);
      alpha[r] = __expf(mrow[r] - nm);
      mrow[r] = nm;
      float sum = 0.0f;
      #pragma unroll
      for (int t = 0; t < 4; ++t) {
        const float p = __expf(s[t][r] - nm);
        s[t][r] = p;
        sum += p;
      }
      sum += __shfl_xor(sum, 1, 32);
      sum += __shfl_xor(sum, 2, 32);
      sum += __shfl_xor(sum, 4, 32);
      sum += __shfl_xor(sum, 8, 32);
      lrow[r] = lrow[r] * alpha[r] + sum;
      #pragma unroll
      for (int t = 0; t < 4; ++t) o[t][r] = o[t][r] * alpha[r];
    }
    // Spill P (16x64, f16, row-major) into this wave's private LDS tile.
    // Same-wave DS ops are in-order, so the A-fragment reads below are safe.
    #pragma unroll
    for (int t = 0; t < 4; ++t)
      #pragma unroll
      for (int r = 0; r < 8; ++r)
        pl[(r + 8 * hi) * 64 + t * 16 + n] = (_Float16)s[t][r];
    // O += P x V   (K-dim = 64 keys, 2 steps of 32)
    #pragma unroll
    for (int ks = 0; ks < 2; ++ks) {
      const half16 aP = load_frag_a(pl + ks * 32, 64);
      #pragma unroll
      for (int t = 0; t < 4; ++t) {
        const half16 bV = load_frag_b(Vl + (t * 16) * 64 + ks * 32, 64);
        o[t] = wmma16(aP, bV, o[t]);
      }
    }
    __syncthreads();   // all waves done with tile before it is overwritten
  }

  float inv[8];
  #pragma unroll
  for (int r = 0; r < 8; ++r) inv[r] = 1.0f / lrow[r];
  #pragma unroll
  for (int t = 0; t < 4; ++t) {
    const int col = h * Hd + t * 16 + n;
    #pragma unroll
    for (int r = 0; r < 8; ++r) {
      const int row = q0 + r + 8 * hi;
      Oh[(size_t)(b * Lsz + row) * Dsz + col] = (_Float16)(o[t][r] * inv[r]);
    }
  }
}

// ---------------------------------------------------------------------------
// Output projection: out = O * Wo^T + bo   (f32 row-major result)
// ---------------------------------------------------------------------------
__global__ __launch_bounds__(256)
void out_gemm_kernel(const _Float16* __restrict__ Ah,
                     const _Float16* __restrict__ Wh,
                     const float* __restrict__ bias,
                     float* __restrict__ out) {
  const int wave = threadIdx.x >> 5;
  const int lane = threadIdx.x & 31;
  const int wm = wave >> 2;
  const int wn = wave & 3;
  const int row0 = blockIdx.y * 128 + wm * 64;
  const int col0 = blockIdx.x * 256 + wn * 64;

  const f32x8 zero = {0.f,0.f,0.f,0.f,0.f,0.f,0.f,0.f};
  f32x8 acc[4][4];
  #pragma unroll
  for (int i = 0; i < 4; ++i)
    #pragma unroll
    for (int j = 0; j < 4; ++j) acc[i][j] = zero;

  for (int k0 = 0; k0 < Dsz; k0 += 32) {
    half16 a[4], bfr[4];
    #pragma unroll
    for (int i = 0; i < 4; ++i)
      a[i] = load_frag_a(Ah + (size_t)(row0 + i * 16) * Dsz + k0, Dsz);
    #pragma unroll
    for (int j = 0; j < 4; ++j)
      bfr[j] = load_frag_b(Wh + (size_t)(col0 + j * 16) * Dsz + k0, Dsz);
    #pragma unroll
    for (int i = 0; i < 4; ++i)
      #pragma unroll
      for (int j = 0; j < 4; ++j)
        acc[i][j] = wmma16(a[i], bfr[j], acc[i][j]);
  }

  const int n  = lane & 15;
  const int hi = lane >> 4;
  #pragma unroll
  for (int j = 0; j < 4; ++j) {
    const int col = col0 + j * 16 + n;
    const float bvv = bias[col];
    #pragma unroll
    for (int i = 0; i < 4; ++i) {
      #pragma unroll
      for (int r = 0; r < 8; ++r) {
        const int row = row0 + i * 16 + r + 8 * hi;
        out[(size_t)row * Dsz + col] = acc[i][j][r] + bvv;
      }
    }
  }
}

extern "C" void kernel_launch(void* const* d_in, const int* in_sizes, int n_in,
                              void* d_out, int out_size, void* d_ws, size_t ws_size,
                              hipStream_t stream) {
  (void)in_sizes; (void)n_in; (void)out_size; (void)ws_size;
  const float* x  = (const float*)d_in[0];
  const float* wq = (const float*)d_in[1];
  const float* bq = (const float*)d_in[2];
  const float* wk = (const float*)d_in[3];
  const float* bk = (const float*)d_in[4];
  const float* wv = (const float*)d_in[5];
  const float* bv = (const float*)d_in[6];
  const float* wo = (const float*)d_in[7];
  const float* bo = (const float*)d_in[8];

  char* ws = (char*)d_ws;
  const size_t MB = 1ull << 20;
  _Float16* Xh  = (_Float16*)(ws + 0);        // 16 MB  (8192 x 1024 f16)
  _Float16* Wqh = (_Float16*)(ws + 16 * MB);  //  2 MB
  _Float16* Wkh = (_Float16*)(ws + 18 * MB);  //  2 MB
  _Float16* Wvh = (_Float16*)(ws + 20 * MB);  //  2 MB
  _Float16* Woh = (_Float16*)(ws + 22 * MB);  //  2 MB
  _Float16* Qh  = (_Float16*)(ws + 24 * MB);  // 16 MB  [b,h,l,hd]
  _Float16* Kh  = (_Float16*)(ws + 40 * MB);  // 16 MB  [b,h,l,hd]
  _Float16* Vt  = (_Float16*)(ws + 56 * MB);  // 16 MB  [b,h,hd,l]
  _Float16* Oh  = (_Float16*)(ws + 72 * MB);  // 16 MB  [b*l, d]

  f32_to_f16_kernel<<<2048, 256, 0, stream>>>(x,  Xh,  Msz * Dsz);
  f32_to_f16_kernel<<<512,  256, 0, stream>>>(wq, Wqh, Dsz * Dsz);
  f32_to_f16_kernel<<<512,  256, 0, stream>>>(wk, Wkh, Dsz * Dsz);
  f32_to_f16_kernel<<<512,  256, 0, stream>>>(wv, Wvh, Dsz * Dsz);
  f32_to_f16_kernel<<<512,  256, 0, stream>>>(wo, Woh, Dsz * Dsz);

  qkv_gemm_kernel<<<dim3(Dsz / 256, Msz / 128, 3), 256, 0, stream>>>(
      Xh, Wqh, Wkh, Wvh, bq, bk, bv, Qh, Kh, Vt);

  flash_attn_kernel<<<dim3(Lsz / 128, Hn, Bsz), 256, 0, stream>>>(Qh, Kh, Vt, Oh);

  out_gemm_kernel<<<dim3(Dsz / 256, Msz / 128, 1), 256, 0, stream>>>(
      Oh, Woh, bo, (float*)d_out);
}
